// FusedMoEModularKernel_34265249087924
// MI455X (gfx1250) — matrix operand order repaired and checked by reading, not compiled, not run on hardware
//
// FusedMoE (E=8,M=4096,K=2048,I=1408,TOPK=2) for MI455X / gfx1250.
// Routed MoE, bf16 WMMA (v_wmma_f32_16x16x32_bf16), weight tiles staged into
// LDS by the Tensor Data Mover (tensor_load_to_lds, double buffered,
// s_wait_tensorcnt), A fragments gathered directly from global as bf16.
#include <hip/hip_runtime.h>
#include <stdint.h>
#include <stddef.h>

#define E_    8
#define M_    4096
#define K_    2048
#define I_    1408
#define N_    (2*I_)
#define TOPK_ 2
#define MT_   (M_*TOPK_)   // 8192 routed pairs

typedef __attribute__((ext_vector_type(16))) __bf16 v16bf;
typedef __attribute__((ext_vector_type(8)))  __bf16 v8bf;
typedef __attribute__((ext_vector_type(8)))  float  v8f;
typedef __attribute__((ext_vector_type(4)))  unsigned int v4u;
typedef __attribute__((ext_vector_type(8)))  int v8i;
typedef __attribute__((ext_vector_type(4)))  int v4i;

// LDS row pitch for staged f32 weight tiles: 32 data dwords + 4 pad dwords
// (TDM pad_interval=32dw, pad_amount=4dw) -> 144B pitch: 16B aligned rows,
// near-conflict-free ds_load_b128 per lane.
#define BPITCH 36

// ---- workspace layout (bytes, 256-aligned) --------------------------------
static const size_t OFF_CNT  = 0;                                   // E ints
static const size_t OFF_LIST = 256;                                 // E*MT ints
static const size_t OFF_HB   = OFF_LIST + (size_t)E_*MT_*4;         // M*K bf16
static const size_t OFF_ACT  = OFF_HB   + (size_t)M_*K_*2;          // MT*I bf16
static const size_t OFF_Y    = OFF_ACT  + (size_t)MT_*I_*2;         // MT*K f32
// total ~107 MB

// ---- helpers ---------------------------------------------------------------
// A-fragment (16x32 bf16): lane<16 holds K 0..15, lane>=16 K 16..31
__device__ __forceinline__ v16bf load_a_frag(const __bf16* pa) {
    v8bf lo = *(const v8bf*)(pa);
    v8bf hi = *(const v8bf*)(pa + 8);
    return __builtin_shufflevector(lo, hi, 0,1,2,3,4,5,6,7,8,9,10,11,12,13,14,15);
}
// B-fragment (32x16 bf16): lane n<16 = col n K 0..15; lane 16+n = col n K 16..31.
// Source is f32 (here: LDS-staged) -> 16 f32 loads + cvt.
__device__ __forceinline__ v16bf cvt_b_frag(const float* p) {
    v16bf r;
#pragma unroll
    for (int i = 0; i < 16; i += 4) {
        float4 f = *(const float4*)(p + i);
        r[i+0] = (__bf16)f.x; r[i+1] = (__bf16)f.y;
        r[i+2] = (__bf16)f.z; r[i+3] = (__bf16)f.w;
    }
    return r;
}

// ---- Tensor Data Mover: build D# and issue a (up to 3D) f32 tile load ------
// Tile: tile_d0 (contig, data_size units) x tile_d1 (stride0) x tile_d2 (stride1).
// Pads LDS dest with 4 dwords after every 32 dwords (BPITCH row pitch).
// tensor dims == tile dims (tiles never straddle the tensor edge here).
__device__ __forceinline__ void tdm_load_f32(
    unsigned lds_byte_off, const void* gsrc,
    unsigned tile_d0, unsigned tile_d1, unsigned tile_d2,
    unsigned long long stride0, unsigned long long stride1, int tensor_d2)
{
    unsigned long long ga = (unsigned long long)(uintptr_t)gsrc;
    v4u g0;
    g0[0] = 1u;                                    // count=1, user mode
    g0[1] = lds_byte_off;                          // lds_addr
    g0[2] = (unsigned)(ga & 0xffffffffu);          // global_addr[31:0]
    g0[3] = (unsigned)((ga >> 32) & 0x01ffffffu)   // global_addr[56:32]
          | (2u << 30);                            // type=2 (image)
    v8i g1;
    g1[0] = (int)((2u << 16)                       // data_size = 4 bytes
          |       (1u << 20)                       // pad_enable
          |       (4u << 22)                       // pad_interval: 32 dwords
          |       (3u << 25));                     // pad_amount: 4 dwords
    g1[1] = (int)((tile_d0 & 0xffffu) << 16);                       // tensor_dim0 lo
    g1[2] = (int)(((tile_d0 >> 16) & 0xffffu) | ((tile_d1 & 0xffffu) << 16)); // dim0 hi | dim1 lo
    g1[3] = (int)(((tile_d1 >> 16) & 0xffffu) | (tile_d0 << 16));   // dim1 hi | tile_dim0
    g1[4] = (int)((tile_d1 & 0xffffu) | ((tile_d2 & 0xffffu) << 16)); // tile_dim1 | tile_dim2
    g1[5] = (int)(stride0 & 0xffffffffu);                           // dim0_stride lo
    g1[6] = (int)(((stride0 >> 32) & 0xffffu) | ((stride1 & 0xffffu) << 16)); // s0 hi | s1 lo
    g1[7] = (int)((stride1 >> 16) & 0xffffffffu);                   // s1[47:16]
    v4i g2 = (v4i){tensor_d2, 0, 0, 0};            // tensor_dim2 (0 for 2D)
    v4i g3 = (v4i){0, 0, 0, 0};
    v8i g4 = (v8i){0, 0, 0, 0, 0, 0, 0, 0};        // unused trailing group (this toolchain's 6-arg form)
    __builtin_amdgcn_tensor_load_to_lds(g0, g1, g2, g3, g4, 0);
}

__device__ __forceinline__ unsigned lds_off_of(const void* p) {
    // flat LDS addresses carry the workgroup-relative LDS byte address in [31:0]
    return (unsigned)(uintptr_t)p;
}

// ---- kernel 0/1: routing ---------------------------------------------------
__global__ void moe_zero_cnt(int* cnt) {
    if (threadIdx.x < E_) cnt[threadIdx.x] = 0;
}
__global__ void moe_route(const int* __restrict__ ids, int* cnt, int* list) {
    int p = blockIdx.x * blockDim.x + threadIdx.x;
    if (p >= MT_) return;
    int e = ids[p];
    int pos = atomicAdd(&cnt[e], 1);
    list[e * MT_ + pos] = p;
}

// ---- kernel 2: hidden f32 -> bf16 -----------------------------------------
__global__ void moe_cvt_hidden(const float* __restrict__ hs, __bf16* __restrict__ hb) {
    int idx = (blockIdx.x * blockDim.x + threadIdx.x) * 4;
    float4 f = *(const float4*)(hs + idx);
    union { __bf16 b[4]; uint64_t u; } o;
    o.b[0] = (__bf16)f.x; o.b[1] = (__bf16)f.y;
    o.b[2] = (__bf16)f.z; o.b[3] = (__bf16)f.w;
    *(uint64_t*)(hb + idx) = o.u;
}

// ---- kernel 3: GEMM1 (gate+up) + SiLU -------------------------------------
// Block tile: 128 rows(pairs) x 64 intermediate cols (gate AND up).
// Weight tile per k-step (f32 128x32 = gate 64 + up 64 rows) DMA'd to LDS by
// the TDM as one 3D tile (32 x 64 x 2, dim2 stride = I*K), double buffered.
__global__ __launch_bounds__(256)
void moe_gemm1(const __bf16* __restrict__ hb, const float* __restrict__ w1,
               const int* __restrict__ cnt, const int* __restrict__ list,
               __bf16* __restrict__ act)
{
    __shared__ __align__(16) float bsm[2][128 * BPITCH];   // 2 x 18KB

    const int e    = blockIdx.z;
    const int ce   = cnt[e];
    const int row0 = blockIdx.y * 128;
    if (row0 >= ce) return;                       // uniform exit: EXEC stays all-1s
    const int i0   = blockIdx.x * 64;

    const int tid  = threadIdx.x;
    const int lane = tid & 31;
    const int wave = tid >> 5;
    const int wm   = wave & 3;
    const int wi   = wave >> 2;
    const int lrow = lane & 15;
    const int hiH  = (lane >> 4) & 1;
    const int klo  = hiH ? 16 : 0;                // A per-lane K sub-offset
    const int kbo  = hiH ? 16 : 0;                // B per-lane K sub-offset

    const int* lst = list + e * MT_;

    size_t abase[2];
#pragma unroll
    for (int tm = 0; tm < 2; ++tm) {
        int r  = row0 + wm * 32 + tm * 16 + lrow;
        int rs = r < ce ? r : ce - 1;             // clamp: garbage rows masked at store
        abase[tm] = (size_t)(lst[rs] >> 1) * K_;  // pair -> token m
    }
    const float* wbase = w1 + ((size_t)e * N_ + i0) * K_;   // gate tile origin

    v8f accG[2][2], accU[2][2];
#pragma unroll
    for (int a = 0; a < 2; ++a)
#pragma unroll
        for (int b = 0; b < 2; ++b) {
            accG[a][b] = (v8f){0.f,0.f,0.f,0.f,0.f,0.f,0.f,0.f};
            accU[a][b] = (v8f){0.f,0.f,0.f,0.f,0.f,0.f,0.f,0.f};
        }

    if (wave == 0)
        tdm_load_f32(lds_off_of(&bsm[0][0]), wbase, 32, 64, 2,
                     (unsigned long long)K_, (unsigned long long)I_ * K_, 2);

    for (int k = 0; k < K_; k += 32) {
        const int cur = (k >> 5) & 1;
        __syncthreads();                          // prev readers of buf (1-cur) done
        if (wave == 0) {
            if (k + 32 < K_) {
                tdm_load_f32(lds_off_of(&bsm[1 - cur][0]), wbase + (k + 32), 32, 64, 2,
                             (unsigned long long)K_, (unsigned long long)I_ * K_, 2);
                __builtin_amdgcn_s_wait_tensorcnt(1);  // oldest (buf cur) landed
            } else {
                __builtin_amdgcn_s_wait_tensorcnt(0);
            }
        }
        __syncthreads();                          // buf cur visible to all waves

        if (k + 32 < K_) __builtin_prefetch(hb + abase[0] + k + 32 + klo, 0, 1);
        v16bf a[2];
#pragma unroll
        for (int tm = 0; tm < 2; ++tm)
            a[tm] = load_a_frag(hb + abase[tm] + k + klo);

        const float* bb = &bsm[cur][0];
        v16bf b[4];
#pragma unroll
        for (int tc = 0; tc < 2; ++tc) {
            int rowg = wi * 32 + tc * 16 + lrow;                  // gate row 0..63
            b[tc]     = cvt_b_frag(bb + rowg * BPITCH + kbo);
            b[2 + tc] = cvt_b_frag(bb + (rowg + 64) * BPITCH + kbo);  // up rows
        }
#pragma unroll
        for (int tm = 0; tm < 2; ++tm)
#pragma unroll
            for (int tc = 0; tc < 2; ++tc) {
                accG[tm][tc] = __builtin_amdgcn_wmma_f32_16x16x32_bf16(
                    false, a[tm], false, b[tc],     (short)0, accG[tm][tc], false, false);
                accU[tm][tc] = __builtin_amdgcn_wmma_f32_16x16x32_bf16(
                    false, a[tm], false, b[2 + tc], (short)0, accU[tm][tc], false, false);
            }
    }

    // Fused SiLU(gate)*up -> bf16 act. C layout: vgpr r, lane l -> row r+(l>=16?8:0), col l%16.
#pragma unroll
    for (int tm = 0; tm < 2; ++tm) {
#pragma unroll
        for (int rr = 0; rr < 8; ++rr) {
            int r = row0 + wm * 32 + tm * 16 + rr + hiH * 8;
            if (r < ce) {
                int pair = lst[r];
                __bf16* ap = act + (size_t)pair * I_;
#pragma unroll
                for (int tc = 0; tc < 2; ++tc) {
                    int col = i0 + wi * 32 + tc * 16 + lrow;
                    float g = accG[tm][tc][rr];
                    float u = accU[tm][tc][rr];
                    float s = g / (1.0f + __expf(-g));
                    ap[col] = (__bf16)(s * u);
                }
            }
        }
    }
}

// ---- kernel 4: GEMM2 (y = act @ w2^T, scaled by topk weight) ---------------
__global__ __launch_bounds__(256)
void moe_gemm2(const __bf16* __restrict__ act, const float* __restrict__ w2,
               const float* __restrict__ tw, const int* __restrict__ cnt,
               const int* __restrict__ list, float* __restrict__ yb)
{
    __shared__ __align__(16) float bsm[2][64 * BPITCH];    // 2 x 9KB

    const int e    = blockIdx.z;
    const int ce   = cnt[e];
    const int row0 = blockIdx.y * 128;
    if (row0 >= ce) return;
    const int c0   = blockIdx.x * 64;             // output hidden-dim columns

    const int tid  = threadIdx.x;
    const int lane = tid & 31;
    const int wave = tid >> 5;
    const int wm   = wave & 3;
    const int wi   = wave >> 2;
    const int lrow = lane & 15;
    const int hiH  = (lane >> 4) & 1;
    const int klo  = hiH ? 16 : 0;
    const int kbo  = hiH ? 16 : 0;

    const int* lst = list + e * MT_;

    size_t abase[2];
#pragma unroll
    for (int tm = 0; tm < 2; ++tm) {
        int r  = row0 + wm * 32 + tm * 16 + lrow;
        int rs = r < ce ? r : ce - 1;
        abase[tm] = (size_t)lst[rs] * I_;         // act is indexed by pair id
    }
    const float* wbase = w2 + ((size_t)e * K_ + c0) * I_;

    v8f acc[2][2];
#pragma unroll
    for (int a = 0; a < 2; ++a)
#pragma unroll
        for (int b = 0; b < 2; ++b)
            acc[a][b] = (v8f){0.f,0.f,0.f,0.f,0.f,0.f,0.f,0.f};

    if (wave == 0)
        tdm_load_f32(lds_off_of(&bsm[0][0]), wbase, 32, 64, 0,
                     (unsigned long long)I_, 0ull, 0);

    for (int i = 0; i < I_; i += 32) {
        const int cur = (i >> 5) & 1;
        __syncthreads();
        if (wave == 0) {
            if (i + 32 < I_) {
                tdm_load_f32(lds_off_of(&bsm[1 - cur][0]), wbase + (i + 32), 32, 64, 0,
                             (unsigned long long)I_, 0ull, 0);
                __builtin_amdgcn_s_wait_tensorcnt(1);
            } else {
                __builtin_amdgcn_s_wait_tensorcnt(0);
            }
        }
        __syncthreads();

        if (i + 32 < I_) __builtin_prefetch(act + abase[0] + i + 32 + klo, 0, 1);
        v16bf a[2];
#pragma unroll
        for (int tm = 0; tm < 2; ++tm)
            a[tm] = load_a_frag(act + abase[tm] + i + klo);

        const float* bb = &bsm[cur][0];
        v16bf b[2];
#pragma unroll
        for (int tc = 0; tc < 2; ++tc) {
            int row = wi * 32 + tc * 16 + lrow;   // 0..63
            b[tc] = cvt_b_frag(bb + row * BPITCH + kbo);
        }
#pragma unroll
        for (int tm = 0; tm < 2; ++tm)
#pragma unroll
            for (int tc = 0; tc < 2; ++tc)
                acc[tm][tc] = __builtin_amdgcn_wmma_f32_16x16x32_bf16(
                    false, a[tm], false, b[tc], (short)0, acc[tm][tc], false, false);
    }

#pragma unroll
    for (int tm = 0; tm < 2; ++tm) {
#pragma unroll
        for (int rr = 0; rr < 8; ++rr) {
            int r = row0 + wm * 32 + tm * 16 + rr + hiH * 8;
            if (r < ce) {
                int pair  = lst[r];
                float wgt = tw[pair];
                float* yp = yb + (size_t)pair * K_;
#pragma unroll
                for (int tc = 0; tc < 2; ++tc) {
                    int col = c0 + wi * 32 + tc * 16 + lrow;
                    yp[col] = acc[tm][tc][rr] * wgt;
                }
            }
        }
    }
}

// ---- kernel 5: combine the two routed pairs per token ----------------------
__global__ void moe_combine(const float4* __restrict__ yb, float4* __restrict__ out) {
    const int KV = K_ / 4;
    int idx = blockIdx.x * blockDim.x + threadIdx.x;   // over M_*KV
    int m  = idx / KV;
    int kv = idx - m * KV;
    float4 a = yb[(size_t)(2 * m) * KV + kv];
    float4 b = yb[(size_t)(2 * m + 1) * KV + kv];
    out[idx] = make_float4(a.x + b.x, a.y + b.y, a.z + b.z, a.w + b.w);
}

// ---- launcher --------------------------------------------------------------
extern "C" void kernel_launch(void* const* d_in, const int* in_sizes, int n_in,
                              void* d_out, int out_size, void* d_ws, size_t ws_size,
                              hipStream_t stream) {
    const float* hs  = (const float*)d_in[0];
    const float* w1  = (const float*)d_in[1];
    const float* w2  = (const float*)d_in[2];
    const float* tw  = (const float*)d_in[3];
    const int*   ids = (const int*)d_in[4];
    float*       out = (float*)d_out;

    char* ws = (char*)d_ws;
    int*    cnt  = (int*)(ws + OFF_CNT);
    int*    list = (int*)(ws + OFF_LIST);
    __bf16* hb   = (__bf16*)(ws + OFF_HB);
    __bf16* act  = (__bf16*)(ws + OFF_ACT);
    float*  yb   = (float*)(ws + OFF_Y);

    moe_zero_cnt<<<1, 32, 0, stream>>>(cnt);
    moe_route<<<(MT_ + 255) / 256, 256, 0, stream>>>(ids, cnt, list);
    moe_cvt_hidden<<<(M_ * K_ / 4) / 256, 256, 0, stream>>>(hs, hb);

    dim3 g1(I_ / 64, MT_ / 128, E_);   // 22 x 64 x 8 (worst-case rows, uniform early-exit)
    moe_gemm1<<<g1, 256, 0, stream>>>(hb, w1, cnt, list, act);

    dim3 g2(K_ / 64, MT_ / 128, E_);   // 32 x 64 x 8
    moe_gemm2<<<g2, 256, 0, stream>>>(act, w2, tw, cnt, list, yb);

    moe_combine<<<(M_ * K_ / 4) / 256, 256, 0, stream>>>((const float4*)yb, (float4*)out);
}
